// ColightModel_48825188221621
// MI455X (gfx1250) — matrix operand order repaired and checked by reading, not compiled, not run on hardware
//
#include <hip/hip_runtime.h>

// ---------------------------------------------------------------------------
// Colight forward for MI455X (gfx1250, wave32, WMMA 16x16x32 f16 -> f32 acc)
// Round 5: fix double-buffer LDS target address in the async-staged GEMM
// (odd K-steps previously computed lds0-4096, an out-of-range LDS write).
// ---------------------------------------------------------------------------
#define R_TOT   131072L
#define OBSD    200
#define EEDGE   4096
#define BBATCH  128

typedef __attribute__((ext_vector_type(16))) _Float16 v16h;
typedef __attribute__((ext_vector_type(8)))  _Float16 v8h;
typedef __attribute__((ext_vector_type(8)))  float    v8f;

// Packed-weight region offsets (in halves), all 16B-aligned.
#define OFF_WQ 0L        // 3 x 128 x 64
#define OFF_WO 24576L    // 3 x  48 x 64
#define OFF_W1 33792L    // 3 x 256 x 64
#define OFF_W2 82944L    // 3 x  48 x 256
#define OFF_E1 119808L   // 128 x 192
#define OFF_E2 144384L   // 128 x 128
#define OFF_WLR 160768L  // 256 x 128 ( [gat_wl ; gat_wr] stacked on N )
#define PACK_TOT 193536L // halves; followed by 256 f32 combined gat bias

__device__ __forceinline__ v8f wmma_f16(v16h a, v16h b, v8f c) {
  return __builtin_amdgcn_wmma_f32_16x16x32_f16(false, a, false, b, (short)0, c,
                                                false, false);
}

__device__ __forceinline__ v16h cat8(v8h lo, v8h hi) {
  return __builtin_shufflevector(lo, hi, 0, 1, 2, 3, 4, 5, 6, 7, 8, 9, 10, 11,
                                 12, 13, 14, 15);
}

// Async 16B global->LDS copy (ASYNCcnt-tracked). LDS address = low 32 bits of
// the generic pointer (ISA 10.2: LDS aperture address = addr[31:0]).
__device__ __forceinline__ void async_copy_b128(unsigned lds_byte_addr,
                                                const void* g) {
  asm volatile("global_load_async_to_lds_b128 %0, %1, off"
               :
               : "v"(lds_byte_addr), "v"(g)
               : "memory");
}
__device__ __forceinline__ void wait_async0() {
  asm volatile("s_wait_asynccnt 0x0" ::: "memory");
}

// A fragment (16x32, MxK) from zero-padded f32 LDS tile (ISA 7.12.2 layout):
// lane l: m=l&15; half i<8 -> K=k0+kh+i, i>=8 -> K=k0+16+kh+(i-8), kh=(l>>4)*8
__device__ __forceinline__ v16h frag_a_lds32(const float* buf, int ld, int m0,
                                             int k0, int lane) {
  const float* p = buf + (m0 + (lane & 15)) * ld + k0 + ((lane >> 4) << 3);
  v16h a;
#pragma unroll
  for (int i = 0; i < 8; ++i) {
    a[i]     = (_Float16)p[i];
    a[i + 8] = (_Float16)p[i + 16];
  }
  return a;
}

// Same from f16 LDS tile: two contiguous 16B runs -> ds_load_b128 pair.
__device__ __forceinline__ v16h frag_a_lds16(const _Float16* buf, int ld,
                                             int m0, int k0, int lane) {
  const _Float16* p = buf + (m0 + (lane & 15)) * ld + k0 + ((lane >> 4) << 3);
  v8h lo = *(const v8h*)p;
  v8h hi = *(const v8h*)(p + 16);
  return cat8(lo, hi);
}

// B fragment (32x16, KxN) for C = X @ W^T from packed f16 W (N x ldk):
// lane l: n=l&15, K runs k0+(l>>4)*8.. and +16 -> two 16B global loads.
__device__ __forceinline__ v16h frag_b_g16(const _Float16* Wp, int ldk, int n0,
                                           int k0, int lane) {
  const _Float16* p = Wp + (long)(n0 + (lane & 15)) * ldk + k0 +
                      ((lane >> 4) << 3);
  v8h lo = *(const v8h*)p;
  v8h hi = *(const v8h*)(p + 16);
  return cat8(lo, hi);
}

// ---------------------------------------------------------------------------
// Kernel 0: pack all GEMM weights to zero-padded f16 + combined gat bias f32.
// ---------------------------------------------------------------------------
__global__ __launch_bounds__(256) void pack_weights_kernel(
    const float* __restrict__ wqkv, const float* __restrict__ wo,
    const float* __restrict__ w1, const float* __restrict__ w2,
    const float* __restrict__ we1, const float* __restrict__ we2,
    const float* __restrict__ wl, const float* __restrict__ wr,
    const float* __restrict__ bl, const float* __restrict__ br,
    _Float16* __restrict__ P, float* __restrict__ BLR) {
  long idx = (long)blockIdx.x * 256 + threadIdx.x;
  if (idx >= PACK_TOT) {  // combined gat bias (256 f32)
    long j = idx - PACK_TOT;
    if (j < 256) BLR[j] = (j < 128) ? bl[j] : br[j - 128];
    return;
  }
  long i = idx;
  float v;
  if (i < OFF_WO) {                    // Wqkv: 3 x (120x40) -> 3 x (128x64)
    long j = i - OFF_WQ;
    int L = (int)(j >> 13), rem = (int)(j & 8191);
    int n = rem >> 6, k = rem & 63;
    v = (n < 120 && k < 40) ? wqkv[L * 4800 + n * 40 + k] : 0.f;
  } else if (i < OFF_W1) {             // Wo: 3 x (40x40) -> 3 x (48x64)
    long j = i - OFF_WO;
    int L = (int)(j / 3072), rem = (int)(j - L * 3072);
    int n = rem >> 6, k = rem & 63;
    v = (n < 40 && k < 40) ? wo[L * 1600 + n * 40 + k] : 0.f;
  } else if (i < OFF_W2) {             // W1: 3 x (256x40) -> 3 x (256x64)
    long j = i - OFF_W1;
    int L = (int)(j >> 14), rem = (int)(j & 16383);
    int n = rem >> 6, k = rem & 63;
    v = (k < 40) ? w1[L * 10240 + n * 40 + k] : 0.f;
  } else if (i < OFF_E1) {             // W2: 3 x (40x256) -> 3 x (48x256)
    long j = i - OFF_W2;
    int L = (int)(j / 12288), rem = (int)(j - L * 12288);
    int n = rem >> 8, k = rem & 255;
    v = (n < 40) ? w2[L * 10240 + n * 256 + k] : 0.f;
  } else if (i < OFF_E2) {             // W_emb1: 128x192 (exact)
    v = we1[i - OFF_E1];
  } else if (i < OFF_WLR) {            // W_emb2: 128x128
    v = we2[i - OFF_E2];
  } else {                             // [gat_wl ; gat_wr]: 256x128
    long j = i - OFF_WLR;
    int n = (int)(j >> 7), k = (int)(j & 127);
    v = (n < 128) ? wl[n * 128 + k] : wr[(n - 128) * 128 + k];
  }
  P[idx] = (_Float16)v;
}

// ---------------------------------------------------------------------------
// Kernel 1: fused 3-layer transformer, 16 rows (64 token-rows) per block.
// 128 threads = 4 waves; wave w owns M-tile w. LDS 63 KB; every GEMM loop
// prefetches the next B fragment while the current WMMAs execute.
// Outputs h and current_obs directly in f16 (consumed as WMMA A operands).
// ---------------------------------------------------------------------------
__global__ __launch_bounds__(128) void tf_kernel(
    const float* __restrict__ obs, const float* __restrict__ pe,
    const _Float16* __restrict__ WqP, const float* __restrict__ bqkv,
    const _Float16* __restrict__ WoP, const float* __restrict__ bo,
    const _Float16* __restrict__ W1P, const float* __restrict__ b1,
    const _Float16* __restrict__ W2P, const float* __restrict__ b2,
    const float* __restrict__ ln1g, const float* __restrict__ ln1b,
    const float* __restrict__ ln2g, const float* __restrict__ ln2b,
    _Float16* __restrict__ h_out, _Float16* __restrict__ cur_out) {
  __shared__ __attribute__((aligned(16))) float    Hs[64 * 64];
  __shared__ __attribute__((aligned(16))) _Float16 QKVs[64 * 120];
  __shared__ __attribute__((aligned(16))) _Float16 FFNs[64 * 256];
  _Float16* const ATTs = FFNs;  // aliased: ATT (64x64) dead before FFN1 writes

  const int tid  = threadIdx.x;
  const int lane = tid & 31;
  const int wave = tid >> 5;
  const long r0  = (long)blockIdx.x * 16;
  const int nl   = lane & 15;
  const int mb   = wave * 16 + (lane >> 4) * 8;

  for (int i = tid; i < 64 * 64; i += 128) {
    int m = i >> 6, d = i & 63;
    int row = m >> 2, t = m & 3;
    Hs[i] = (d < 40) ? obs[(r0 + row) * OBSD + t * 40 + d] + pe[t * 40 + d]
                     : 0.f;
  }
  for (int i = tid; i < 16 * 32; i += 128) {
    int row = i >> 5, c = i & 31;
    cur_out[(r0 + row) * 32 + c] = (_Float16)obs[(r0 + row) * OBSD + 160 + c];
  }
  __syncthreads();

  for (int L = 0; L < 3; ++L) {
    const _Float16* Wq = WqP + L * 8192;   // 128x64
    const _Float16* Wo = WoP + L * 3072;   // 48x64
    const _Float16* W1 = W1P + L * 16384;  // 256x64
    const _Float16* W2 = W2P + L * 12288;  // 48x256
    const float* Bq = bqkv + L * 120;
    const float* Bo = bo + L * 40;
    const float* B1 = b1 + L * 256;
    const float* B2 = b2 + L * 40;
    const float* G1 = ln1g + L * 40;  const float* Bb1 = ln1b + L * 40;
    const float* G2 = ln2g + L * 40;  const float* Bb2 = ln2b + L * 40;

    // --- QKV = Hs(64x64p) @ Wq^T + Bq -> QKVs (f16, ld 120) ---------------
    {
      v16h a0 = frag_a_lds32(Hs, 64, wave * 16, 0, lane);
      v16h a1 = frag_a_lds32(Hs, 64, wave * 16, 32, lane);
      v16h b0 = frag_b_g16(Wq, 64, 0, 0, lane);
      v16h b1 = frag_b_g16(Wq, 64, 0, 32, lane);
      for (int nt = 0; nt < 8; ++nt) {
        int ntn = (nt < 7) ? nt + 1 : 7;
        v16h p0 = frag_b_g16(Wq, 64, ntn * 16, 0, lane);
        v16h p1 = frag_b_g16(Wq, 64, ntn * 16, 32, lane);
        int nn = nt * 16 + nl;
        float bb = Bq[nn < 120 ? nn : 119];
        v8f acc = {};
        acc = wmma_f16(a0, b0, acc);
        acc = wmma_f16(a1, b1, acc);
        if (nn < 120) {
#pragma unroll
          for (int rr = 0; rr < 8; ++rr)
            QKVs[(mb + rr) * 120 + nn] = (_Float16)(acc[rr] + bb);
        }
        b0 = p0;
        b1 = p1;
      }
    }
    __syncthreads();

    // --- causal attention (seq 4, 2 heads x 20); thread = (row,tq,head) ---
    {
      int row = tid >> 3, tq = (tid >> 1) & 3, head = tid & 1;
      int mq = row * 4 + tq;
      float q[20];
#pragma unroll
      for (int d = 0; d < 20; ++d) q[d] = (float)QKVs[mq * 120 + head * 20 + d];
      float sc[4];
      float mx = -1e30f;
      for (int tk = 0; tk < 4; ++tk) {
        float s = -1e30f;
        if (tk <= tq) {
          s = 0.f;
#pragma unroll
          for (int d = 0; d < 20; ++d)
            s += q[d] * (float)QKVs[(row * 4 + tk) * 120 + 40 + head * 20 + d];
          s *= 0.22360680f;  // 1/sqrt(20)
        }
        sc[tk] = s;
        mx = fmaxf(mx, s);
      }
      float dsum = 0.f;
      for (int tk = 0; tk < 4; ++tk) {
        float e = (tk <= tq) ? __expf(sc[tk] - mx) : 0.f;
        sc[tk] = e;
        dsum += e;
      }
      float inv = 1.f / dsum;
      float o[20];
#pragma unroll
      for (int d = 0; d < 20; ++d) o[d] = 0.f;
      for (int tk = 0; tk < 4; ++tk) {
        float wgt = sc[tk] * inv;
#pragma unroll
        for (int d = 0; d < 20; ++d)
          o[d] += wgt * (float)QKVs[(row * 4 + tk) * 120 + 80 + head * 20 + d];
      }
#pragma unroll
      for (int d = 0; d < 20; ++d)
        ATTs[mq * 64 + head * 20 + d] = (_Float16)o[d];
      for (int i = tid; i < 64 * 24; i += 128) {  // zero K-pad cols 40..63
        int m = i / 24, d = 40 + (i - m * 24);
        ATTs[m * 64 + d] = (_Float16)0.f;
      }
    }
    __syncthreads();

    // --- O = ATT(64x64p) @ Wo^T + Bo -> QKVs[:, 0:40] scratch -------------
    {
      v16h a0 = frag_a_lds16(ATTs, 64, wave * 16, 0, lane);
      v16h a1 = frag_a_lds16(ATTs, 64, wave * 16, 32, lane);
      v16h b0 = frag_b_g16(Wo, 64, 0, 0, lane);
      v16h b1 = frag_b_g16(Wo, 64, 0, 32, lane);
#pragma unroll
      for (int nt = 0; nt < 3; ++nt) {
        int ntn = (nt < 2) ? nt + 1 : 2;
        v16h p0 = frag_b_g16(Wo, 64, ntn * 16, 0, lane);
        v16h p1 = frag_b_g16(Wo, 64, ntn * 16, 32, lane);
        int nn = nt * 16 + nl;
        float bb = Bo[nn < 40 ? nn : 39];
        v8f acc = {};
        acc = wmma_f16(a0, b0, acc);
        acc = wmma_f16(a1, b1, acc);
        if (nn < 40) {
#pragma unroll
          for (int rr = 0; rr < 8; ++rr)
            QKVs[(mb + rr) * 120 + nn] = (_Float16)(acc[rr] + bb);
        }
        b0 = p0;
        b1 = p1;
      }
    }
    __syncthreads();

    // --- h = LN1(h + O) ---------------------------------------------------
    if (tid < 64) {
      float x[40];
      float s = 0.f;
#pragma unroll
      for (int d = 0; d < 40; ++d) {
        x[d] = Hs[tid * 64 + d] + (float)QKVs[tid * 120 + d];
        s += x[d];
      }
      float mean = s * (1.0f / 40.0f);
      float v = 0.f;
#pragma unroll
      for (int d = 0; d < 40; ++d) { float t = x[d] - mean; v += t * t; }
      float inv = rsqrtf(v * (1.0f / 40.0f) + 1e-6f);
#pragma unroll
      for (int d = 0; d < 40; ++d)
        Hs[tid * 64 + d] = (x[d] - mean) * inv * G1[d] + Bb1[d];
    }
    __syncthreads();

    // --- FFN1 = relu(h @ W1^T + B1) -> FFNs (f16, 64x256) -----------------
    {
      v16h a0 = frag_a_lds32(Hs, 64, wave * 16, 0, lane);
      v16h a1 = frag_a_lds32(Hs, 64, wave * 16, 32, lane);
      v16h b0 = frag_b_g16(W1, 64, 0, 0, lane);
      v16h b1 = frag_b_g16(W1, 64, 0, 32, lane);
      for (int nt = 0; nt < 16; ++nt) {
        int ntn = (nt < 15) ? nt + 1 : 15;
        v16h p0 = frag_b_g16(W1, 64, ntn * 16, 0, lane);
        v16h p1 = frag_b_g16(W1, 64, ntn * 16, 32, lane);
        int nn = nt * 16 + nl;
        float bb = B1[nn];
        v8f acc = {};
        acc = wmma_f16(a0, b0, acc);
        acc = wmma_f16(a1, b1, acc);
#pragma unroll
        for (int rr = 0; rr < 8; ++rr)
          FFNs[(mb + rr) * 256 + nn] = (_Float16)fmaxf(acc[rr] + bb, 0.f);
        b0 = p0;
        b1 = p1;
      }
    }
    __syncthreads();

    // --- FFN2 = FFN(64x256) @ W2^T + B2 -> QKVs[:, 0:40] ------------------
    {
#pragma unroll
      for (int nt = 0; nt < 3; ++nt) {
        v16h b = frag_b_g16(W2, 256, nt * 16, 0, lane);
        v8f acc = {};
        for (int kt = 0; kt < 8; ++kt) {
          int ktn = (kt < 7) ? kt + 1 : 7;
          v16h nb = frag_b_g16(W2, 256, nt * 16, ktn * 32, lane);
          v16h a = frag_a_lds16(FFNs, 256, wave * 16, kt * 32, lane);
          acc = wmma_f16(a, b, acc);
          b = nb;
        }
        int nn = nt * 16 + nl;
        float bb = B2[nn < 40 ? nn : 39];
        if (nn < 40) {
#pragma unroll
          for (int rr = 0; rr < 8; ++rr)
            QKVs[(mb + rr) * 120 + nn] = (_Float16)(acc[rr] + bb);
        }
      }
    }
    __syncthreads();

    // --- h = LN2(h + FFN2) ------------------------------------------------
    if (tid < 64) {
      float x[40];
      float s = 0.f;
#pragma unroll
      for (int d = 0; d < 40; ++d) {
        x[d] = Hs[tid * 64 + d] + (float)QKVs[tid * 120 + d];
        s += x[d];
      }
      float mean = s * (1.0f / 40.0f);
      float v = 0.f;
#pragma unroll
      for (int d = 0; d < 40; ++d) { float t = x[d] - mean; v += t * t; }
      float inv = rsqrtf(v * (1.0f / 40.0f) + 1e-6f);
#pragma unroll
      for (int d = 0; d < 40; ++d)
        Hs[tid * 64 + d] = (x[d] - mean) * inv * G2[d] + Bb2[d];
    }
    __syncthreads();
  }

  for (int i = tid; i < 64 * 40; i += 128) {
    int m = i / 40, d = i - m * 40;
    int row = m >> 2, t = m & 3;
    h_out[(r0 + row) * 160 + t * 40 + d] = (_Float16)Hs[m * 64 + d];
  }
}

// ---------------------------------------------------------------------------
// Kernel 2: C = act(A @ W^T + bias). A is f16 (concat of A1|A2 on K; the
// split at K1 is K-step aligned so each 32-K step has a uniform source).
// BM=64 x BN=128 per 256-thread block; wave = one N-tile (one B frag per
// K-step reused by 4 WMMAs). A tile staged by async global->LDS b128 copies
// (1 x 16B per thread per K-step), double-buffered, 1 barrier per K-step.
// C written as f16 (feeds next GEMM) or f32 (feeds GAT scatter).
// ---------------------------------------------------------------------------
__global__ __launch_bounds__(256) void gemm_kernel(
    const _Float16* __restrict__ A1, int K1, const _Float16* __restrict__ A2,
    const _Float16* __restrict__ Wp, const float* __restrict__ bias,
    float* __restrict__ C32, _Float16* __restrict__ C16, int K, int N,
    int relu) {
  __shared__ __attribute__((aligned(16))) _Float16 As[2][64 * 32];
  const int tid  = threadIdx.x;
  const int lane = tid & 31;
  const int wave = tid >> 5;
  const long m0  = (long)blockIdx.x * 64;
  const int n0   = blockIdx.y * 128 + wave * 16;
  const int nk   = K >> 5;
  const int Krem = K - K1;
  // this thread's 16B copy slot: row crow, 8-half column block ccol
  const int crow = tid >> 2;
  const int ccol = (tid & 3) * 8;
  const unsigned ldsA = (unsigned)(unsigned long long)&As[0][0] + tid * 16u;
  const unsigned ldsB = ldsA + 4096u;  // As[1] slot

  // prologue: async-stage K-step 0 into buffer 0
  async_copy_b128(ldsA, A1 + (m0 + crow) * (long)K1 + ccol);  // k0=0 < K1
  v16h b = frag_b_g16(Wp, K, n0, 0, lane);
  v8f acc0 = {}, acc1 = {}, acc2 = {}, acc3 = {};
  wait_async0();
  __syncthreads();

  for (int ks = 0; ks < nk; ++ks) {
    const _Float16* Ab = As[ks & 1];
    int k0n = ((ks + 1 < nk) ? ks + 1 : ks) << 5;
    // async prefetch of next A tile into the other buffer + next B frag
    {
      const _Float16* g = (k0n < K1)
          ? A1 + (m0 + crow) * (long)K1 + k0n + ccol
          : A2 + (m0 + crow) * (long)Krem + (k0n - K1) + ccol;
      async_copy_b128(((ks + 1) & 1) ? ldsB : ldsA, g);
    }
    v16h nb = frag_b_g16(Wp, K, n0, k0n, lane);
    v16h a0 = frag_a_lds16(Ab, 32, 0, 0, lane);
    v16h a1 = frag_a_lds16(Ab, 32, 16, 0, lane);
    v16h a2 = frag_a_lds16(Ab, 32, 32, 0, lane);
    v16h a3 = frag_a_lds16(Ab, 32, 48, 0, lane);
    acc0 = wmma_f16(a0, b, acc0);
    acc1 = wmma_f16(a1, b, acc1);
    acc2 = wmma_f16(a2, b, acc2);
    acc3 = wmma_f16(a3, b, acc3);
    b = nb;
    wait_async0();
    __syncthreads();
  }

  const int n = n0 + (lane & 15);
  const float bb = bias[n];
  const long mbase = m0 + (lane >> 4) * 8;
#pragma unroll
  for (int t = 0; t < 4; ++t) {
    v8f acc = (t == 0) ? acc0 : (t == 1) ? acc1 : (t == 2) ? acc2 : acc3;
    long mb = mbase + t * 16;
#pragma unroll
    for (int rr = 0; rr < 8; ++rr) {
      float v = acc[rr] + bb;
      if (relu) v = fmaxf(v, 0.f);
      if (C16) C16[(mb + rr) * (long)N + n] = (_Float16)v;
      else     C32[(mb + rr) * (long)N + n] = v;
    }
  }
}

// ---------------------------------------------------------------------------
__global__ __launch_bounds__(256) void zero_kernel(float* __restrict__ p,
                                                   long n) {
  long i = (long)blockIdx.x * 256 + threadIdx.x;
  if (i < n) p[i] = 0.f;
}

// ---------------------------------------------------------------------------
// GAT edge scatter; one thread per (edge, head). xlr is interleaved (R,256):
// cols 0..127 = xl, 128..255 = xr. Softmax shift-invariance lets us drop
// segment_max exactly (logits are O(0.05)-scale; no overflow risk).
// ---------------------------------------------------------------------------
__global__ __launch_bounds__(256) void gat_edge_kernel(
    const int* __restrict__ edge_index, const float* __restrict__ xlr,
    const float* __restrict__ att, float* __restrict__ den,
    float* __restrict__ outacc, long TOT) {
  long t = (long)blockIdx.x * 256 + threadIdx.x;
  if (t >= TOT * 4) return;
  long eid = t >> 2;
  int h = (int)(t & 3);
  const long BE = (long)BBATCH * EEDGE;
  long src, dst;
  if (eid < BE) {
    long b = eid >> 12;
    int j = (int)(eid & 4095);
    src = (long)edge_index[j] + (b << 10);
    dst = (long)edge_index[EEDGE + j] + (b << 10);
  } else {
    src = dst = eid - BE;
  }
  const float4* pl = (const float4*)(xlr + src * 256 + h * 32);
  const float4* pr = (const float4*)(xlr + dst * 256 + 128 + h * 32);
  const float4* pa = (const float4*)(att + h * 32);
  float xs[32];
  float e = 0.f;
#pragma unroll
  for (int v = 0; v < 8; ++v) {
    float4 lv = pl[v], rv = pr[v], av = pa[v];
    float m0 = lv.x + rv.x, m1 = lv.y + rv.y, m2 = lv.z + rv.z,
          m3 = lv.w + rv.w;
    m0 = (m0 > 0.f) ? m0 : 0.2f * m0;
    m1 = (m1 > 0.f) ? m1 : 0.2f * m1;
    m2 = (m2 > 0.f) ? m2 : 0.2f * m2;
    m3 = (m3 > 0.f) ? m3 : 0.2f * m3;
    e += av.x * m0 + av.y * m1 + av.z * m2 + av.w * m3;
    xs[v * 4 + 0] = lv.x;
    xs[v * 4 + 1] = lv.y;
    xs[v * 4 + 2] = lv.z;
    xs[v * 4 + 3] = lv.w;
  }
  float w = __expf(e);
  atomicAdd(&den[dst * 4 + h], w);
  float* po = outacc + dst * 128 + h * 32;
#pragma unroll
  for (int c = 0; c < 32; ++c) atomicAdd(&po[c], w * xs[c]);
}

// ---------------------------------------------------------------------------
// Normalize GAT output + bias, apply head W_q (8x128). Thread per (row, a).
// ---------------------------------------------------------------------------
__global__ __launch_bounds__(256) void final_kernel(
    const float* __restrict__ outacc, const float* __restrict__ den,
    const float* __restrict__ gbias, const float* __restrict__ Wq,
    const float* __restrict__ bq, float* __restrict__ out) {
  long t = (long)blockIdx.x * 256 + threadIdx.x;
  if (t >= R_TOT * 8) return;
  long r = t >> 3;
  int a = (int)(t & 7);
  const float4* oa = (const float4*)(outacc + r * 128);
  const float4* gb = (const float4*)gbias;
  const float4* w4 = (const float4*)(Wq + a * 128);
  const float* dn = den + r * 4;
  float inv[4] = {1.f / dn[0], 1.f / dn[1], 1.f / dn[2], 1.f / dn[3]};
  float s = bq[a];
#pragma unroll
  for (int v = 0; v < 32; ++v) {
    float4 o = oa[v], g = gb[v], ww = w4[v];
    float iv = inv[v >> 3];
    s += ww.x * (o.x * iv + g.x) + ww.y * (o.y * iv + g.y) +
         ww.z * (o.z * iv + g.z) + ww.w * (o.w * iv + g.w);
  }
  out[t] = s;
}

// ---------------------------------------------------------------------------
extern "C" void kernel_launch(void* const* d_in, const int* in_sizes, int n_in,
                              void* d_out, int out_size, void* d_ws,
                              size_t ws_size, hipStream_t stream) {
  (void)in_sizes; (void)n_in; (void)out_size; (void)ws_size;
  const float* obs      = (const float*)d_in[0];
  const int*   eidx     = (const int*)d_in[1];
  const float* W_emb1   = (const float*)d_in[2];
  const float* b_emb1   = (const float*)d_in[3];
  const float* W_emb2   = (const float*)d_in[4];
  const float* b_emb2   = (const float*)d_in[5];
  const float* tf_wqkv  = (const float*)d_in[6];
  const float* tf_bqkv  = (const float*)d_in[7];
  const float* tf_wo    = (const float*)d_in[8];
  const float* tf_bo    = (const float*)d_in[9];
  const float* ffn_w1   = (const float*)d_in[10];
  const float* ffn_b1   = (const float*)d_in[11];
  const float* ffn_w2   = (const float*)d_in[12];
  const float* ffn_b2   = (const float*)d_in[13];
  const float* ln1_g    = (const float*)d_in[14];
  const float* ln1_b    = (const float*)d_in[15];
  const float* ln2_g    = (const float*)d_in[16];
  const float* ln2_b    = (const float*)d_in[17];
  const float* gat_wl   = (const float*)d_in[18];
  const float* gat_bl   = (const float*)d_in[19];
  const float* gat_wr   = (const float*)d_in[20];
  const float* gat_br   = (const float*)d_in[21];
  const float* gat_att  = (const float*)d_in[22];
  const float* gat_bias = (const float*)d_in[23];
  const float* W_q      = (const float*)d_in[24];
  const float* b_q      = (const float*)d_in[25];
  const float* pe       = (const float*)d_in[26];

  float* ws = (float*)d_ws;
  const long R = R_TOT;
  // f16 activation chain + f32 GAT buffers, lifetime-overlapped (~204 MB):
  _Float16* h16   = (_Float16*)ws;              // R*160 halves [0, R*80)
  _Float16* cur16 = (_Float16*)(ws + R * 80);   // R*32 halves  [R*80, R*96)
  _Float16* Y116  = (_Float16*)(ws + R * 96);   // R*128 halves [R*96, R*160)
  _Float16* emb16 = (_Float16*)(ws + R * 256);  // R*128 halves [R*256, R*320)
  float* xlr      = ws;                         // f32 (R,256)  [0, R*256)
  float* outacc   = ws + R * 256;               // f32 (R,128), overlays emb16
  float* den      = ws + R * 384;               // f32 (R,4)
  _Float16* Pk    = (_Float16*)(ws + R * 388);  // packed f16 weights
  float* BLR      = (float*)(Pk + PACK_TOT);    // combined gat bias (256)
  float* out      = (float*)d_out;

  // 0) pack weights (+ combined gat bias)
  pack_weights_kernel<<<dim3((unsigned)((PACK_TOT + 256 + 255) / 256)),
                        dim3(256), 0, stream>>>(
      tf_wqkv, tf_wo, ffn_w1, ffn_w2, W_emb1, W_emb2, gat_wl, gat_wr, gat_bl,
      gat_br, Pk, BLR);

  // 1) fused transformer + current_obs extraction (f16 outputs)
  tf_kernel<<<dim3((unsigned)(R / 16)), dim3(128), 0, stream>>>(
      obs, pe, Pk + OFF_WQ, tf_bqkv, Pk + OFF_WO, tf_bo, Pk + OFF_W1, ffn_b1,
      Pk + OFF_W2, ffn_b2, ln1_g, ln1_b, ln2_g, ln2_b, h16, cur16);

  // 2) Y1 = relu(concat(h, cur) @ W_emb1^T + b): K=192 (split 160|32), N=128
  gemm_kernel<<<dim3((unsigned)(R / 64), 1), dim3(256), 0, stream>>>(
      h16, 160, cur16, Pk + OFF_E1, b_emb1, nullptr, Y116, 192, 128, 1);
  // 3) emb = relu(Y1 @ W_emb2^T + b): K=128, N=128
  gemm_kernel<<<dim3((unsigned)(R / 64), 1), dim3(256), 0, stream>>>(
      Y116, 128, Y116, Pk + OFF_E2, b_emb2, nullptr, emb16, 128, 128, 1);
  // 4) xlr = emb @ [gat_wl;gat_wr]^T + [bl;br]: K=128, N=256, f32 out
  gemm_kernel<<<dim3((unsigned)(R / 64), 2), dim3(256), 0, stream>>>(
      emb16, 128, emb16, Pk + OFF_WLR, BLR, xlr, nullptr, 128, 256, 0);

  // 5) zero accumulators (outacc overlays emb16, dead now)
  zero_kernel<<<dim3((unsigned)((R * 128 + 255) / 256)), dim3(256), 0,
                stream>>>(outacc, R * 128);
  zero_kernel<<<dim3((unsigned)((R * 4 + 255) / 256)), dim3(256), 0, stream>>>(
      den, R * 4);

  // 6) edge scatter
  const long TOT = (long)BBATCH * EEDGE + R;  // 655360
  gat_edge_kernel<<<dim3((unsigned)((TOT * 4 + 255) / 256)), dim3(256), 0,
                    stream>>>(eidx, xlr, gat_att, den, outacc, TOT);

  // 7) normalize + head
  final_kernel<<<dim3((unsigned)((R * 8 + 255) / 256)), dim3(256), 0, stream>>>(
      outacc, den, gat_bias, W_q, b_q, out);
}
